// GNN_44220983279812
// MI455X (gfx1250) — matrix operand order repaired and checked by reading, not compiled, not run on hardware
//
#include <hip/hip_runtime.h>

#define N_NODES 1024
#define N_EDGES (N_NODES * N_NODES)
#define H 4
#define C 32
#define HC 128
#define NEG_SLOPE 0.2f

typedef _Float16 v8h  __attribute__((ext_vector_type(8)));
typedef _Float16 v16h __attribute__((ext_vector_type(16)));
typedef float    v8f  __attribute__((ext_vector_type(8)));

#define CAT16(lo, hi) __builtin_shufflevector((lo), (hi), 0,1,2,3,4,5,6,7,8,9,10,11,12,13,14,15)

// ---------- CDNA5 async memory->LDS copy (ASYNCcnt-tracked) ----------
// LDS generic addresses carry the LDS byte offset in their low 32 bits.
__device__ __forceinline__ void async_copy_b128(const void* gptr, void* lptr) {
    unsigned      ldsoff = (unsigned)(size_t)lptr;
    unsigned long long ga = (unsigned long long)(size_t)gptr;
    asm volatile("global_load_async_to_lds_b128 %0, %1, off"
                 :: "v"(ldsoff), "v"(ga)
                 : "memory");
}
__device__ __forceinline__ void wait_async0() {
    asm volatile("s_wait_asynccnt 0" ::: "memory");
}

// ---------- monotonic float ordering for atomicMax on unsigned ----------
__device__ __forceinline__ unsigned encOrd(float f) {
    unsigned u = __float_as_uint(f);
    return (u & 0x80000000u) ? ~u : (u | 0x80000000u);
}
__device__ __forceinline__ float decOrd(unsigned u) {
    return (u & 0x80000000u) ? __uint_as_float(u ^ 0x80000000u) : __uint_as_float(~u);
}

// ---------------------------- utility fills ----------------------------
__global__ void k_fill_f32(float* p, float v, int n) {
    int i = blockIdx.x * blockDim.x + threadIdx.x;
    if (i < n) p[i] = v;
}
__global__ void k_fill_u32(unsigned* p, unsigned v, int n) {
    int i = blockIdx.x * blockDim.x + threadIdx.x;
    if (i < n) p[i] = v;
}

// k1[h] = sum_c lin_edge1[h*32+c] * att_edge1[h*32+c]
__global__ void k_prep_edge(const float* __restrict__ lin_e,
                            const float* __restrict__ att_e,
                            float* __restrict__ k1) {
    __shared__ float s[HC];
    int j = threadIdx.x;
    s[j] = lin_e[j] * att_e[j];
    __syncthreads();
    if (j < H) {
        float a = 0.f;
        for (int c = 0; c < C; ++c) a += s[j * C + c];
        k1[j] = a;
    }
}

// h = x @ W  ([N,Din] @ [Din,128]) ; a_s/a_d = per-head attention reductions
__global__ void k_node_transform(const float* __restrict__ xin, int Din,
                                 const float* __restrict__ W,
                                 const float* __restrict__ att_s,
                                 const float* __restrict__ att_d,
                                 float* __restrict__ hout,
                                 float* __restrict__ a_s,
                                 float* __restrict__ a_d) {
    int n = blockIdx.x;
    int j = threadIdx.x;          // 0..127
    float acc = 0.f;
    for (int d = 0; d < Din; ++d) acc += xin[n * Din + d] * W[d * HC + j];
    hout[n * HC + j] = acc;
    __shared__ float ss[HC], sd[HC];
    ss[j] = acc * att_s[j];
    sd[j] = acc * att_d[j];
    __syncthreads();
    if (j < H) {
        float s1 = 0.f, s2 = 0.f;
        for (int c = 0; c < C; ++c) { s1 += ss[j * C + c]; s2 += sd[j * C + c]; }
        a_s[n * H + j] = s1;
        a_d[n * H + j] = s2;
    }
}

// scatter-mean stats for self-loop edge attributes
__global__ void k_loopstats(const int* __restrict__ dst, const float* __restrict__ ef,
                            float* __restrict__ lsum, float* __restrict__ lcnt, int E) {
    int i = blockIdx.x * blockDim.x + threadIdx.x;
    if (i < E) {
        int d = dst[i];
        atomicAdd(&lsum[d], ef[i]);
        atomicAdd(&lcnt[d], 1.0f);
    }
}

// pass 1: alpha = leakyrelu(a_s[src] + a_d[dst] (+ edge term)); segment max
__global__ void k_alpha_max(const int* __restrict__ src, const int* __restrict__ dst,
                            const float* __restrict__ ef, const float* __restrict__ k1,
                            const float* __restrict__ lsum, const float* __restrict__ lcnt,
                            const float* __restrict__ a_s, const float* __restrict__ a_d,
                            float* __restrict__ alpha, unsigned* __restrict__ amax,
                            int E, int Nn, int use_edge) {
    int i = blockIdx.x * blockDim.x + threadIdx.x;
    if (i >= E + Nn) return;
    int s, d;
    float et = 0.f;
    if (i < E) {
        s = src[i]; d = dst[i];
        if (use_edge) et = ef[i];
    } else {
        s = d = i - E;
        if (use_edge) {
            float c = lcnt[s];
            et = lsum[s] / fmaxf(c, 1.0f);
        }
    }
    #pragma unroll
    for (int h = 0; h < H; ++h) {
        float a = a_s[s * H + h] + a_d[d * H + h];
        if (use_edge) a += et * k1[h];
        a = (a > 0.f) ? a : a * NEG_SLOPE;
        alpha[(size_t)i * H + h] = a;
        atomicMax(&amax[d * H + h], encOrd(a));
    }
}

// pass 2: ex = exp(alpha - amax[dst]); segment sum
__global__ void k_alpha_exp(const int* __restrict__ dst, const unsigned* __restrict__ amax,
                            float* __restrict__ alpha, float* __restrict__ denom,
                            int E, int Nn) {
    int i = blockIdx.x * blockDim.x + threadIdx.x;
    if (i >= E + Nn) return;
    int d = (i < E) ? dst[i] : (i - E);
    #pragma unroll
    for (int h = 0; h < H; ++h) {
        float m  = decOrd(amax[d * H + h]);
        float ex = __expf(alpha[(size_t)i * H + h] - m);
        alpha[(size_t)i * H + h] = ex;
        atomicAdd(&denom[d * H + h], ex);
    }
}

// pass 3: agg[dst] += w * h[src]  (one block of 128 per edge, j = h*32+c)
__global__ void k_aggregate(const int* __restrict__ src, const int* __restrict__ dst,
                            const float* __restrict__ alpha, const float* __restrict__ denom,
                            const float* __restrict__ hbuf, float* __restrict__ agg, int E) {
    int i = blockIdx.x;
    int j = threadIdx.x;            // 0..127
    int s, d;
    if (i < E) { s = src[i]; d = dst[i]; }
    else       { s = d = i - E; }
    int hh = j >> 5;
    float w = alpha[(size_t)i * H + hh] / (denom[d * H + hh] + 1e-16f);
    atomicAdd(&agg[d * HC + j], w * hbuf[s * HC + j]);
}

// head mean + bias, plus f16 copy for WMMA consumption
__global__ void k_finalize(const float* __restrict__ agg, const float* __restrict__ bias,
                           float* __restrict__ xo, _Float16* __restrict__ xoh) {
    int n = blockIdx.x, c = threadIdx.x;   // 32 threads
    float v = 0.25f * (agg[n * HC + c] + agg[n * HC + 32 + c] +
                       agg[n * HC + 64 + c] + agg[n * HC + 96 + c]) + bias[c];
    xo[n * C + c]  = v;
    xoh[n * C + c] = (_Float16)v;
}

// W[K][N] f32  ->  WT[N][K] f16  (transposed for contiguous B fragments)
__global__ void k_convw(const float* __restrict__ W, _Float16* __restrict__ WT,
                        int K, int Ncol) {
    int i = blockIdx.x * blockDim.x + threadIdx.x;
    if (i < K * Ncol) {
        int k = i / Ncol, n = i % Ncol;
        WT[n * K + k] = (_Float16)W[i];
    }
}

// ------------------------- edge MLP (WMMA core) -------------------------
// 128 threads (4 waves), 64 edges per block.
// FC1: [64,128]f16 @ [128,64] -> relu -> f16
// FC2: [64,64]f16  @ [64,32]  -> relu -> f32
// FC3: [64,32]f32  @ [32,2]   -> softmax (VALU)
union BufA {
    _Float16 feats[64 * 128];   // 16 KB
    float    h2[64 * 32];       // 8 KB (aliases; feats dead after FC1)
};

__global__ void __launch_bounds__(128)
k_edge_mlp(const int* __restrict__ src, const int* __restrict__ dst,
           const _Float16* __restrict__ x0h, const _Float16* __restrict__ x1h,
           const _Float16* __restrict__ w1T, const _Float16* __restrict__ w2T,
           const float* __restrict__ b1, const float* __restrict__ b2,
           const float* __restrict__ w3, const float* __restrict__ b3,
           float* __restrict__ out) {
    __shared__ alignas(16) _Float16 sW1[64 * 128];   // 16 KB, [n][k]
    __shared__ alignas(16) _Float16 sW2[32 * 64];    // 4 KB,  [n][k]
    __shared__ alignas(16) _Float16 sH1[64 * 64];    // 8 KB
    __shared__ alignas(16) BufA bufA;

    const int t    = threadIdx.x;
    const int lane = t & 31;
    const int wave = t >> 5;
    const int e0   = blockIdx.x * 64;

    __builtin_prefetch(w3, 0, 0);   // global_prefetch_b8: tiny FC3 weights into cache

    // stage pre-transposed f16 weights into LDS: async memory->LDS (b128 chunks)
    {
        #pragma unroll
        for (int i = t; i < (64 * 128) / 8; i += 128)
            async_copy_b128(w1T + i * 8, &sW1[i * 8]);
        #pragma unroll
        for (int i = t; i < (32 * 64) / 8; i += 128)
            async_copy_b128(w2T + i * 8, &sW2[i * 8]);
    }

    // gather [x0[s], x0[d], x1[s], x1[d]] -> feats[et][0..127] via async-to-LDS
    {
        const int et = t >> 1;
        const int e  = e0 + et;
        const int s  = src[e];
        const int d  = dst[e];
        const _Float16* xs = (t & 1) ? x1h : x0h;
        const int base     = (t & 1) ? 64 : 0;
        #pragma unroll
        for (int q = 0; q < 4; ++q)
            async_copy_b128(xs + s * C + q * 8, &bufA.feats[et * 128 + base + q * 8]);
        #pragma unroll
        for (int q = 0; q < 4; ++q)
            async_copy_b128(xs + d * C + q * 8, &bufA.feats[et * 128 + base + 32 + q * 8]);
    }
    wait_async0();
    __syncthreads();

    // per-lane fragment addressing (CDNA5 16-bit A/B VGPR layouts)
    const int mrow = wave * 16 + (lane & 15);       // A row
    const int akb  = (lane < 16) ? 0 : 8;           // A: K base of first 8 halves
    const int nb   = lane & 15;                     // B column within N-tile
    const int bkb  = (lane < 16) ? 0 : 8;           // B: K base of first 8 halves
    const int rr   = wave * 16 + ((lane >= 16) ? 8 : 0);  // C-layout row base

    // ---------------- FC1: 4 N-tiles x 4 K-steps ----------------
    // Hoist the 4 A-fragments once; reuse across all N-tiles.
    v16h aF1[4];
    #pragma unroll
    for (int kt = 0; kt < 4; ++kt) {
        const int k0 = kt * 32;
        v8h alo = *(const v8h*)(&bufA.feats[mrow * 128 + k0 + akb]);
        v8h ahi = *(const v8h*)(&bufA.feats[mrow * 128 + k0 + akb + 16]);
        aF1[kt] = CAT16(alo, ahi);
    }
    #pragma unroll
    for (int nt = 0; nt < 4; ++nt) {
        v8f acc = {0.f, 0.f, 0.f, 0.f, 0.f, 0.f, 0.f, 0.f};
        #pragma unroll
        for (int kt = 0; kt < 4; ++kt) {
            const int k0 = kt * 32;
            const _Float16* bp = &sW1[(nt * 16 + nb) * 128 + k0 + bkb];
            v8h blo = *(const v8h*)bp;
            v8h bhi = *(const v8h*)(bp + 16);
            v16h b  = CAT16(blo, bhi);
            acc = __builtin_amdgcn_wmma_f32_16x16x32_f16(false, aF1[kt], false, b,
                                                         (short)0, acc, false, false);
        }
        const float bias = b1[nt * 16 + nb];
        #pragma unroll
        for (int r = 0; r < 8; ++r) {
            float v = acc[r] + bias;
            v = (v > 0.f) ? v : 0.f;
            sH1[(rr + r) * 64 + nt * 16 + nb] = (_Float16)v;
        }
    }
    __syncthreads();   // sH1 complete; feats now dead -> h2 may alias it

    // ---------------- FC2: 2 N-tiles x 2 K-steps ----------------
    v16h aF2[2];
    #pragma unroll
    for (int kt = 0; kt < 2; ++kt) {
        const int k0 = kt * 32;
        v8h alo = *(const v8h*)(&sH1[mrow * 64 + k0 + akb]);
        v8h ahi = *(const v8h*)(&sH1[mrow * 64 + k0 + akb + 16]);
        aF2[kt] = CAT16(alo, ahi);
    }
    #pragma unroll
    for (int nt = 0; nt < 2; ++nt) {
        v8f acc = {0.f, 0.f, 0.f, 0.f, 0.f, 0.f, 0.f, 0.f};
        #pragma unroll
        for (int kt = 0; kt < 2; ++kt) {
            const int k0 = kt * 32;
            const _Float16* bp = &sW2[(nt * 16 + nb) * 64 + k0 + bkb];
            v8h blo = *(const v8h*)bp;
            v8h bhi = *(const v8h*)(bp + 16);
            v16h b  = CAT16(blo, bhi);
            acc = __builtin_amdgcn_wmma_f32_16x16x32_f16(false, aF2[kt], false, b,
                                                         (short)0, acc, false, false);
        }
        const float bias = b2[nt * 16 + nb];
        #pragma unroll
        for (int r = 0; r < 8; ++r) {
            float v = acc[r] + bias;
            v = (v > 0.f) ? v : 0.f;
            bufA.h2[(rr + r) * 32 + nt * 16 + nb] = v;
        }
    }
    __syncthreads();

    // ---------------- FC3 + softmax (VALU) ----------------
    if (t < 64) {
        const float* h2row = &bufA.h2[t * 32];
        float z0 = b3[0], z1 = b3[1];
        #pragma unroll
        for (int c = 0; c < 32; ++c) {
            float hv = h2row[c];
            z0 += hv * w3[c * 2 + 0];
            z1 += hv * w3[c * 2 + 1];
        }
        float m   = fmaxf(z0, z1);
        float p0  = __expf(z0 - m);
        float p1  = __expf(z1 - m);
        float inv = 1.0f / (p0 + p1);
        size_t e = (size_t)(e0 + t);
        out[e * 2 + 0] = p0 * inv;
        out[e * 2 + 1] = p1 * inv;
    }
}

// ------------------------------- launcher -------------------------------
extern "C" void kernel_launch(void* const* d_in, const int* in_sizes, int n_in,
                              void* d_out, int out_size, void* d_ws, size_t ws_size,
                              hipStream_t stream) {
    (void)in_sizes; (void)n_in; (void)out_size; (void)ws_size;
    const float* x      = (const float*)d_in[0];
    const int*   edges  = (const int*)d_in[1];
    const float* ef     = (const float*)d_in[2];
    const float* W1     = (const float*)d_in[3];
    const float* att_s1 = (const float*)d_in[4];
    const float* att_d1 = (const float*)d_in[5];
    const float* lin_e1 = (const float*)d_in[6];
    const float* att_e1 = (const float*)d_in[7];
    const float* bias1  = (const float*)d_in[8];
    const float* W2     = (const float*)d_in[9];
    const float* att_s2 = (const float*)d_in[10];
    const float* att_d2 = (const float*)d_in[11];
    const float* bias2  = (const float*)d_in[12];
    const float* fc1W   = (const float*)d_in[13];
    const float* fc1b   = (const float*)d_in[14];
    const float* fc2W   = (const float*)d_in[15];
    const float* fc2b   = (const float*)d_in[16];
    const float* fc3W   = (const float*)d_in[17];
    const float* fc3b   = (const float*)d_in[18];
    float* out = (float*)d_out;

    const int* srcI = edges;
    const int* dstI = edges + N_EDGES;

    char* ws = (char*)d_ws;
    size_t off = 0;
    auto take = [&](size_t bytes) -> char* {
        char* p = ws + off;
        off = (off + bytes + 255) & ~(size_t)255;
        return p;
    };
    float*     hbuf  = (float*)take((size_t)N_NODES * HC * 4);
    float*     as    = (float*)take((size_t)N_NODES * H * 4);
    float*     ad    = (float*)take((size_t)N_NODES * H * 4);
    float*     k1    = (float*)take(256);
    float*     lsum  = (float*)take((size_t)N_NODES * 4);
    float*     lcnt  = (float*)take((size_t)N_NODES * 4);
    float*     alpha = (float*)take((size_t)(N_EDGES + N_NODES) * H * 4);
    unsigned*  amax  = (unsigned*)take((size_t)N_NODES * H * 4);
    float*     denom = (float*)take((size_t)N_NODES * H * 4);
    float*     agg   = (float*)take((size_t)N_NODES * HC * 4);
    float*     x0    = (float*)take((size_t)N_NODES * C * 4);
    _Float16*  x0h   = (_Float16*)take((size_t)N_NODES * C * 2);
    float*     x1    = (float*)take((size_t)N_NODES * C * 4);
    _Float16*  x1h   = (_Float16*)take((size_t)N_NODES * C * 2);
    _Float16*  w1T   = (_Float16*)take((size_t)64 * 128 * 2);
    _Float16*  w2T   = (_Float16*)take((size_t)32 * 64 * 2);

    const int EN = N_EDGES + N_NODES;
    const dim3 b256(256);
    const unsigned NH  = N_NODES * H;
    const unsigned NHC = N_NODES * HC;

    // pre-transpose/convert edge-MLP weights to f16
    k_convw<<<dim3((128 * 64 + 255) / 256), b256, 0, stream>>>(fc1W, w1T, 128, 64);
    k_convw<<<dim3((64 * 32 + 255) / 256),  b256, 0, stream>>>(fc2W, w2T, 64, 32);

    // ---------------- GAT layer 1 (with edge attrs) ----------------
    k_fill_u32<<<dim3((NH + 255) / 256), b256, 0, stream>>>(amax, 0u, NH);
    k_fill_f32<<<dim3((NH + 255) / 256), b256, 0, stream>>>(denom, 0.f, NH);
    k_fill_f32<<<dim3((NHC + 255) / 256), b256, 0, stream>>>(agg, 0.f, NHC);
    k_fill_f32<<<dim3((N_NODES + 255) / 256), b256, 0, stream>>>(lsum, 0.f, N_NODES);
    k_fill_f32<<<dim3((N_NODES + 255) / 256), b256, 0, stream>>>(lcnt, 0.f, N_NODES);
    k_prep_edge<<<1, 128, 0, stream>>>(lin_e1, att_e1, k1);
    k_node_transform<<<N_NODES, 128, 0, stream>>>(x, 3, W1, att_s1, att_d1, hbuf, as, ad);
    k_loopstats<<<dim3((N_EDGES + 255) / 256), b256, 0, stream>>>(dstI, ef, lsum, lcnt, N_EDGES);
    k_alpha_max<<<dim3((EN + 255) / 256), b256, 0, stream>>>(srcI, dstI, ef, k1, lsum, lcnt,
                                                             as, ad, alpha, amax,
                                                             N_EDGES, N_NODES, 1);
    k_alpha_exp<<<dim3((EN + 255) / 256), b256, 0, stream>>>(dstI, amax, alpha, denom,
                                                             N_EDGES, N_NODES);
    k_aggregate<<<dim3(EN), dim3(128), 0, stream>>>(srcI, dstI, alpha, denom, hbuf, agg, N_EDGES);
    k_finalize<<<dim3(N_NODES), dim3(32), 0, stream>>>(agg, bias1, x0, x0h);

    // ---------------- GAT layer 2 (no edge attrs) ----------------
    k_fill_u32<<<dim3((NH + 255) / 256), b256, 0, stream>>>(amax, 0u, NH);
    k_fill_f32<<<dim3((NH + 255) / 256), b256, 0, stream>>>(denom, 0.f, NH);
    k_fill_f32<<<dim3((NHC + 255) / 256), b256, 0, stream>>>(agg, 0.f, NHC);
    k_node_transform<<<N_NODES, 128, 0, stream>>>(x0, 32, W2, att_s2, att_d2, hbuf, as, ad);
    k_alpha_max<<<dim3((EN + 255) / 256), b256, 0, stream>>>(srcI, dstI, ef, k1, lsum, lcnt,
                                                             as, ad, alpha, amax,
                                                             N_EDGES, N_NODES, 0);
    k_alpha_exp<<<dim3((EN + 255) / 256), b256, 0, stream>>>(dstI, amax, alpha, denom,
                                                             N_EDGES, N_NODES);
    k_aggregate<<<dim3(EN), dim3(128), 0, stream>>>(srcI, dstI, alpha, denom, hbuf, agg, N_EDGES);
    k_finalize<<<dim3(N_NODES), dim3(32), 0, stream>>>(agg, bias2, x1, x1h);

    // ---------------- edge MLP (WMMA) ----------------
    k_edge_mlp<<<dim3(N_EDGES / 64), dim3(128), 0, stream>>>(srcI, dstI, x0h, x1h,
                                                             w1T, w2T, fc1b, fc2b,
                                                             fc3W, fc3b, out);
}
